// CausalSelfAttention_31250182046312
// MI455X (gfx1250) — matrix operand order repaired and checked by reading, not compiled, not run on hardware
//
#include <hip/hip_runtime.h>
#include <hip/hip_bf16.h>

// ---------------------------------------------------------------------------
// CausalSelfAttention forward for MI455X (gfx1250, wave32, WMMA).
// Pipeline:
//   1) convert x -> bf16; transpose-convert Wqkv, Wout -> bf16 [N][K]
//   2) QKV GEMM (bf16 WMMA, fp32 acc) fused with RoPE; writes Q,K [B,H,T,DH]
//      bf16 (Q pre-scaled by 1/sqrt(DH)) and V transposed [B,H,DH,T] bf16
//   3) flash attention (online softmax); K/V tiles staged into LDS once per
//      block via double-buffered GLOBAL_LOAD_ASYNC_TO_LDS_B128 (ASYNCcnt)
//   4) out-projection GEMM -> fp32 d_out
// ---------------------------------------------------------------------------

typedef __bf16 bf16_t;
typedef __attribute__((ext_vector_type(16))) __bf16 v16bf;
typedef __attribute__((ext_vector_type(8)))  __bf16 v8bf;
typedef __attribute__((ext_vector_type(4)))  __bf16 v4bf;
typedef __attribute__((ext_vector_type(8)))  float  v8f;

namespace {
constexpr int kB  = 4;
constexpr int kT  = 2048;
constexpr int kD  = 1024;
constexpr int kH  = 16;
constexpr int kDH = 64;
constexpr int kM  = kB * kT;     // 8192 rows of x
constexpr int kNq = 3 * kD;      // 3072 qkv columns
}

__device__ __forceinline__ v8f wmma_bf16(v16bf a, v16bf b, v8f c) {
  return __builtin_amdgcn_wmma_f32_16x16x32_bf16(false, a, false, b, (short)0, c,
                                                 false, false);
}

__device__ __forceinline__ v16bf cat8(v8bf lo, v8bf hi) {
  return __builtin_shufflevector(lo, hi, 0, 1, 2, 3, 4, 5, 6, 7, 8, 9, 10, 11,
                                 12, 13, 14, 15);
}

// A fragment: 16x32 bf16 tile of row-major A, rows row0..row0+15, k0..k0+31.
// Lane l: m = l&15, half = l>>4. VGPRs 0-3 hold K = 8*half..8*half+7 (packed
// pairs), VGPRs 4-7 hold K = 16+8*half..16+8*half+7  (ISA 7.12.2).
__device__ __forceinline__ v16bf load_a_frag(const bf16_t* __restrict__ A,
                                             int lda, int row0, int k0,
                                             int lane) {
  const int m = lane & 15, half = lane >> 4;
  const bf16_t* p = A + (size_t)(row0 + m) * lda + k0 + 8 * half;
  return cat8(*reinterpret_cast<const v8bf*>(p),
              *reinterpret_cast<const v8bf*>(p + 16));
}

// B fragment: 32x16 bf16 tile of B, read from the TRANSPOSED operand
// Bt[n][k] (row-major, leading dim ldk).  Lane l: n = l&15, half = l>>4 and
// the lane holds K = 8*half..8*half+7 and 16+8*half..16+8*half+7 (as A).
// Works for both global and LDS-resident Bt (compiler emits ds_load for AS3).
__device__ __forceinline__ v16bf load_b_frag(const bf16_t* Bt, int ldk, int n0,
                                             int k0, int lane) {
  const int n = lane & 15, half = lane >> 4;
  const bf16_t* p = Bt + (size_t)(n0 + n) * ldk + k0 + 8 * half;
  return cat8(*reinterpret_cast<const v8bf*>(p),
              *reinterpret_cast<const v8bf*>(p + 16));
}

// Async DMA of one 16-byte chunk, global -> LDS (tracked by ASYNCcnt).
// Low 32 bits of a generic pointer to a __shared__ object are the LDS offset
// (ISA 10.2: LDS aperture address = addr[31:0]).
__device__ __forceinline__ void async_g2l_b128(const void* gptr, void* lptr) {
  unsigned lds = (unsigned)(unsigned long long)(uintptr_t)lptr;
  unsigned long long g = (unsigned long long)(uintptr_t)gptr;
  asm volatile("global_load_async_to_lds_b128 %0, %1, off"
               :: "v"(lds), "v"(g) : "memory");
}

// ---------------------------------------------------------------------------
__global__ void csa_cvt_bf16(const float* __restrict__ src,
                             bf16_t* __restrict__ dst, int n4) {
  int i = blockIdx.x * blockDim.x + threadIdx.x;
  if (i < n4) {
    float4 v = reinterpret_cast<const float4*>(src)[i];
    v4bf o = {(bf16_t)v.x, (bf16_t)v.y, (bf16_t)v.z, (bf16_t)v.w};
    reinterpret_cast<v4bf*>(dst)[i] = o;
  }
}

// src[K][N] fp32  ->  dst[N][K] bf16  (transposed for B-fragment loads)
__global__ void csa_cvt_transpose(const float* __restrict__ src,
                                  bf16_t* __restrict__ dst, int K, int N) {
  int idx = blockIdx.x * blockDim.x + threadIdx.x;
  if (idx < K * N) {
    int n = idx / K;
    int k = idx - n * K;
    dst[idx] = (bf16_t)src[(size_t)k * N + n];
  }
}

// ---------------------------------------------------------------------------
// QKV = Xb @ Wqkv, fused with RoPE + scatter.
// grid (kM/128, kNq/64), block 128 (4 waves). Wave tile: 32 rows x 64 cols.
__global__ void __launch_bounds__(128)
csa_gemm_qkv_rope(const bf16_t* __restrict__ Xb, const bf16_t* __restrict__ WT,
                  const float* __restrict__ cosT, const float* __restrict__ sinT,
                  bf16_t* __restrict__ Qo, bf16_t* __restrict__ Ko,
                  bf16_t* __restrict__ VTo) {
  const int lane = threadIdx.x & 31;
  const int wave = threadIdx.x >> 5;
  const int row0 = blockIdx.x * 128 + wave * 32;
  const int n0   = blockIdx.y * 64;

  v8f acc[2][4] = {};
  for (int k0 = 0; k0 < kD; k0 += 32) {
    __builtin_prefetch(Xb + (size_t)(row0 + (lane & 15)) * kD + k0 + 64, 0, 1);
    v16bf a0 = load_a_frag(Xb, kD, row0, k0, lane);
    v16bf a1 = load_a_frag(Xb, kD, row0 + 16, k0, lane);
#pragma unroll
    for (int j = 0; j < 4; ++j) {
      v16bf b = load_b_frag(WT, kD, n0 + 16 * j, k0, lane);
      acc[0][j] = wmma_bf16(a0, b, acc[0][j]);
      acc[1][j] = wmma_bf16(a1, b, acc[1][j]);
    }
  }

  // C-layout: lane holds column n = lane&15; vgpr v holds row v + 8*(lane>>4).
  const int n = lane & 15, g = lane >> 4;
  const int region = n0 >> 10;            // 0 = Q, 1 = K, 2 = V
  const int h = (n0 & (kD - 1)) >> 6;     // head (tile is head-aligned, 64 cols)

  if (region < 2) {
    bf16_t* dst = (region == 0) ? Qo : Ko;
    const float sc = (region == 0) ? 0.125f : 1.0f;   // 1/sqrt(DH) folded into Q
#pragma unroll
    for (int mi = 0; mi < 2; ++mi) {
#pragma unroll
      for (int v = 0; v < 8; ++v) {
        int row = row0 + 16 * mi + v + 8 * g;
        int bb = row >> 11, t = row & (kT - 1);
        const float* cp = cosT + t * kDH;
        const float* sp = sinT + t * kDH;
        size_t base = (((size_t)bb * kH + h) * kT + t) * kDH;
#pragma unroll
        for (int j = 0; j < 2; ++j) {     // frag j pairs with frag j+2 (dh+32)
          int dh = 16 * j + n;
          float x1 = acc[mi][j][v];
          float x2 = acc[mi][j + 2][v];
          float r1 = (x1 * cp[dh] - x2 * sp[dh]) * sc;
          float r2 = (x2 * cp[dh + 32] + x1 * sp[dh + 32]) * sc;
          dst[base + dh]      = (bf16_t)r1;
          dst[base + dh + 32] = (bf16_t)r2;
        }
      }
    }
  } else {  // V: store transposed [B,H,DH,T] so PV B-fragments load contiguous
#pragma unroll
    for (int mi = 0; mi < 2; ++mi) {
#pragma unroll
      for (int v = 0; v < 8; ++v) {
        int row = row0 + 16 * mi + v + 8 * g;
        int bb = row >> 11, t = row & (kT - 1);
        size_t hb = ((size_t)bb * kH + h) * kDH;
#pragma unroll
        for (int j = 0; j < 4; ++j) {
          int dh = 16 * j + n;
          VTo[(hb + dh) * kT + t] = (bf16_t)acc[mi][j][v];
        }
      }
    }
  }
}

// ---------------------------------------------------------------------------
// Flash attention: grid (T/64, B*H), block 128 (4 waves, 16 q-rows per wave).
// K/V tiles are staged into LDS once per block with double-buffered async DMA
// (GLOBAL_LOAD_ASYNC_TO_LDS_B128), overlapping the next tile's DMA with the
// current tile's WMMA work.  All 4 waves share the staged tiles via ds loads.
__global__ void __launch_bounds__(128)
csa_flash_attn(const bf16_t* __restrict__ Qi, const bf16_t* __restrict__ Ki,
               const bf16_t* __restrict__ VTi, bf16_t* __restrict__ AO) {
  __shared__ __align__(16) bf16_t ktile[2][32][64];  // [kv rows][dh]     8 KB
  __shared__ __align__(16) bf16_t vtile[2][64][32];  // [dh rows][kv]     8 KB
  __shared__ __align__(16) bf16_t pbuf[4][16][48];   // P transpose tiles 6 KB

  const int tid = threadIdx.x;
  const int lane = tid & 31;
  const int wave = tid >> 5;
  const int bh = blockIdx.y;
  const int q0 = blockIdx.x * 64 + wave * 16;        // this wave's 16 q rows
  const bf16_t* qb = Qi + (size_t)bh * kT * kDH;
  const bf16_t* kb = Ki + (size_t)bh * kT * kDH;
  const bf16_t* vb = VTi + (size_t)bh * kDH * kT;
  const int n = lane & 15, g = lane >> 4;

  // Stage one KV tile (kv0..kv0+31) into LDS buffer bufi: 512 x b128 chunks
  // spread over 128 threads (4 async DMAs per thread).
  auto stage = [&](int kv0, int bufi) {
#pragma unroll
    for (int c = tid; c < 256; c += 128) {   // K tile: 32 rows x 128 B
      const char* gp = (const char*)(kb + (size_t)(kv0 + (c >> 3)) * kDH) +
                       (c & 7) * 16;
      async_g2l_b128(gp, (char*)&ktile[bufi][0][0] + c * 16);
    }
#pragma unroll
    for (int c = tid; c < 256; c += 128) {   // V tile: 64 rows x 64 B
      const char* gp = (const char*)(vb + (size_t)(c >> 2) * kT + kv0) +
                       (c & 3) * 16;
      async_g2l_b128(gp, (char*)&vtile[bufi][0][0] + c * 16);
    }
  };

  const v16bf qa0 = load_a_frag(qb, kDH, q0, 0, lane);
  const v16bf qa1 = load_a_frag(qb, kDH, q0, 32, lane);

  v8f o0 = {}, o1 = {}, o2 = {}, o3 = {};
  float mrow[8], lrow[8];
#pragma unroll
  for (int v = 0; v < 8; ++v) { mrow[v] = -3.0e38f; lrow[v] = 0.0f; }

  // Unified trip count across the block (required for shared staging): run to
  // the last tile needed by wave 3; the elementwise mask neutralizes tiles
  // past a wave's own diagonal (exp(-3e38 - m) == 0 under online softmax).
  const int kv_end = blockIdx.x * 64 + 32;

  stage(0, 0);
  asm volatile("s_wait_asynccnt 0x0" ::: "memory");
  __syncthreads();

  for (int kv0 = 0; kv0 <= kv_end; kv0 += 32) {
    const int buf = (kv0 >> 5) & 1;
    if (kv0 + 32 <= kv_end) stage(kv0 + 32, buf ^ 1);   // overlap next DMA

    // S = (Q*scale) @ K^T from the LDS K tile (ds_load_b128 fragments).
    const bf16_t* kt = &ktile[buf][0][0];
    v8f s0 = {}, s1 = {};
    s0 = wmma_bf16(qa0, load_b_frag(kt, 64, 0, 0, lane), s0);
    s0 = wmma_bf16(qa1, load_b_frag(kt, 64, 0, 32, lane), s0);
    s1 = wmma_bf16(qa0, load_b_frag(kt, 64, 16, 0, lane), s1);
    s1 = wmma_bf16(qa1, load_b_frag(kt, 64, 16, 32, lane), s1);

    if (kv0 + 31 > q0) {                    // causal mask (diag + beyond)
#pragma unroll
      for (int v = 0; v < 8; ++v) {
        int row = q0 + v + 8 * g;
        if (kv0 + n > row)      s0[v] = -3.0e38f;
        if (kv0 + 16 + n > row) s1[v] = -3.0e38f;
      }
    }

    // Online softmax: each (vgpr, lane-group) is one row; reduce over 16 lanes.
#pragma unroll
    for (int v = 0; v < 8; ++v) {
      float mx = fmaxf(s0[v], s1[v]);
#pragma unroll
      for (int off = 1; off < 16; off <<= 1)
        mx = fmaxf(mx, __shfl_xor(mx, off, 32));
      float mnew = fmaxf(mrow[v], mx);
      float corr = __expf(mrow[v] - mnew);
      float p0 = __expf(s0[v] - mnew);
      float p1 = __expf(s1[v] - mnew);
      s0[v] = p0; s1[v] = p1;
      float rs = p0 + p1;
#pragma unroll
      for (int off = 1; off < 16; off <<= 1) rs += __shfl_xor(rs, off, 32);
      lrow[v] = lrow[v] * corr + rs;
      mrow[v] = mnew;
      o0[v] *= corr; o1[v] *= corr; o2[v] *= corr; o3[v] *= corr;
    }

    // Transpose P from C-layout to A-layout through per-wave LDS tile.
#pragma unroll
    for (int v = 0; v < 8; ++v) {
      pbuf[wave][v + 8 * g][n]      = (bf16_t)s0[v];
      pbuf[wave][v + 8 * g][16 + n] = (bf16_t)s1[v];
    }
    asm volatile("s_wait_dscnt 0x0" ::: "memory");  // cross-lane LDS RAW
    const bf16_t* pp = &pbuf[wave][lane & 15][8 * (lane >> 4)];
    v16bf pa = cat8(*reinterpret_cast<const v8bf*>(pp),
                    *reinterpret_cast<const v8bf*>(pp + 16));

    // O += P @ V from the LDS V tile.
    const bf16_t* vt = &vtile[buf][0][0];
    o0 = wmma_bf16(pa, load_b_frag(vt, 32, 0, 0, lane), o0);
    o1 = wmma_bf16(pa, load_b_frag(vt, 32, 16, 0, lane), o1);
    o2 = wmma_bf16(pa, load_b_frag(vt, 32, 32, 0, lane), o2);
    o3 = wmma_bf16(pa, load_b_frag(vt, 32, 48, 0, lane), o3);

    // Next tile's DMA must be complete and all waves done with this buffer
    // before it is overwritten in the following iteration.
    asm volatile("s_wait_asynccnt 0x0" ::: "memory");
    __syncthreads();
  }

  const int bb = bh >> 4, h = bh & 15;
#pragma unroll
  for (int v = 0; v < 8; ++v) {
    int t = q0 + v + 8 * g;
    float inv = 1.0f / lrow[v];
    size_t base = ((size_t)bb * kT + t) * kD + h * kDH;
    AO[base + n]      = (bf16_t)(o0[v] * inv);
    AO[base + 16 + n] = (bf16_t)(o1[v] * inv);
    AO[base + 32 + n] = (bf16_t)(o2[v] * inv);
    AO[base + 48 + n] = (bf16_t)(o3[v] * inv);
  }
}

// ---------------------------------------------------------------------------
// out = AO @ Wout  (fp32 result). grid (kM/128, kD/64), block 128.
__global__ void __launch_bounds__(128)
csa_gemm_out(const bf16_t* __restrict__ Ab, const bf16_t* __restrict__ WT,
             float* __restrict__ C) {
  const int lane = threadIdx.x & 31;
  const int wave = threadIdx.x >> 5;
  const int row0 = blockIdx.x * 128 + wave * 32;
  const int n0   = blockIdx.y * 64;

  v8f acc[2][4] = {};
  for (int k0 = 0; k0 < kD; k0 += 32) {
    __builtin_prefetch(Ab + (size_t)(row0 + (lane & 15)) * kD + k0 + 64, 0, 1);
    v16bf a0 = load_a_frag(Ab, kD, row0, k0, lane);
    v16bf a1 = load_a_frag(Ab, kD, row0 + 16, k0, lane);
#pragma unroll
    for (int j = 0; j < 4; ++j) {
      v16bf b = load_b_frag(WT, kD, n0 + 16 * j, k0, lane);
      acc[0][j] = wmma_bf16(a0, b, acc[0][j]);
      acc[1][j] = wmma_bf16(a1, b, acc[1][j]);
    }
  }
  const int n = lane & 15, g = lane >> 4;
#pragma unroll
  for (int mi = 0; mi < 2; ++mi)
#pragma unroll
    for (int v = 0; v < 8; ++v) {
      size_t row = row0 + 16 * mi + v + 8 * g;
#pragma unroll
      for (int j = 0; j < 4; ++j)
        C[row * kD + n0 + 16 * j + n] = acc[mi][j][v];
    }
}

// ---------------------------------------------------------------------------
extern "C" void kernel_launch(void* const* d_in, const int* in_sizes, int n_in,
                              void* d_out, int out_size, void* d_ws,
                              size_t ws_size, hipStream_t stream) {
  (void)in_sizes; (void)n_in; (void)out_size; (void)ws_size;
  const float* x    = (const float*)d_in[0];
  const float* cosT = (const float*)d_in[1];   // [T, DH] flat
  const float* sinT = (const float*)d_in[2];
  const float* Wqkv = (const float*)d_in[3];   // [D, 3D]
  const float* Wout = (const float*)d_in[4];   // [D, D]
  float* out = (float*)d_out;

  // Workspace layout (bytes), total ~88 MB of bf16 intermediates.
  char* ws = (char*)d_ws;
  bf16_t* Xb  = (bf16_t*)(ws + 0);          // [8192,1024]       16 MB
  bf16_t* WqT = (bf16_t*)(ws + 16777216);   // [3072,1024]        6 MB
  bf16_t* WoT = (bf16_t*)(ws + 23068672);   // [1024,1024]        2 MB
  bf16_t* Qb  = (bf16_t*)(ws + 25165824);   // [B,H,T,DH]        16 MB
  bf16_t* Kb  = (bf16_t*)(ws + 41943040);   // [B,H,T,DH]        16 MB
  bf16_t* VTb = (bf16_t*)(ws + 58720256);   // [B,H,DH,T]        16 MB
  bf16_t* AOb = (bf16_t*)(ws + 75497472);   // [8192,1024]       16 MB

  csa_cvt_bf16<<<(kM * kD / 4 + 255) / 256, 256, 0, stream>>>(x, Xb, kM * kD / 4);
  csa_cvt_transpose<<<(kD * kNq + 255) / 256, 256, 0, stream>>>(Wqkv, WqT, kD, kNq);
  csa_cvt_transpose<<<(kD * kD + 255) / 256, 256, 0, stream>>>(Wout, WoT, kD, kD);

  csa_gemm_qkv_rope<<<dim3(kM / 128, kNq / 64), 128, 0, stream>>>(
      Xb, WqT, cosT, sinT, Qb, Kb, VTb);

  csa_flash_attn<<<dim3(kT / 64, kB * kH), 128, 0, stream>>>(Qb, Kb, VTb, AOb);

  csa_gemm_out<<<dim3(kM / 128, kD / 64), 128, 0, stream>>>(AOb, WoT, out);
}
